// SpectralSelfAttention_19035295056174
// MI455X (gfx1250) — matrix operand order repaired and checked by reading, not compiled, not run on hardware
//
#include <hip/hip_runtime.h>
#include <hip/hip_bf16.h>
#include <math.h>

// ---------------- CDNA5 (gfx1250) types ----------------
typedef __attribute__((ext_vector_type(16))) __bf16 v16bf;
typedef __attribute__((ext_vector_type(8)))  float  v8f;
typedef __attribute__((ext_vector_type(4)))  unsigned int v4u;
typedef __attribute__((ext_vector_type(8)))  int v8i_;
typedef __attribute__((ext_vector_type(4)))  int v4i_;
typedef unsigned short u16t;

#define HWN   4096
#define CIN   256
#define CH_   128
#define ROWS  32768        // B*HW

// workspace byte offsets (all 16B aligned)
#define OFF_SIG   0
#define OFF_WCAT  256                                  // swizzled f|g|h weights: 96 blk *512 elt
#define OFF_WO    (256 + 96*512*2)                     // swizzled wo: 64 blk * 512
#define OFF_FG    (OFF_WO + 64*512*2)                  // fg [ROWS][64] bf16
#define OFF_HT    (OFF_FG + (size_t)ROWS*64*2)         // hT [8][128][4096] bf16
#define OFF_OBF   (OFF_HT + (size_t)8*128*4096*2)      // o  [ROWS][128] bf16

__device__ __forceinline__ __bf16 f2bf(float f) { return (__bf16)f; }   // native RTNE cvt
__device__ __forceinline__ unsigned short f2bfbits(float f) {
    __bf16 h = (__bf16)f;
    unsigned short s; __builtin_memcpy(&s, &h, 2);
    return s;
}

__device__ __forceinline__ v16bf load_frag32(const u16t* p) {   // 32B contiguous, 16B aligned
    v16bf r;
    uint4 a0 = *(const uint4*)p;
    uint4 a1 = *(const uint4*)(p + 8);
    __builtin_memcpy(&r, &a0, 16);
    __builtin_memcpy(((char*)&r) + 16, &a1, 16);
    return r;
}

// ---- Tensor Data Mover: 2D tile load, data_size=2B (CDNA5 ISA D# groups 0/1) ----
// pad_interval_code: LDS padding every (2<<code) DWORDs; pad_amount_code: (code+1) DWORDs
__device__ __forceinline__ void tdm_load_2d(unsigned lds_addr, const void* gptr,
                                            unsigned tile_d0, unsigned tile_d1,
                                            unsigned tensor_d0, unsigned tensor_d1,
                                            unsigned stride_d0,
                                            unsigned pad_en,
                                            unsigned pad_interval_code,
                                            unsigned pad_amount_code) {
    unsigned long long ga = (unsigned long long)(size_t)gptr;
    v4u g0;
    g0[0] = 1u;                                          // count=1, user mode
    g0[1] = lds_addr;                                    // lds_addr [63:32]
    g0[2] = (unsigned)(ga & 0xffffffffu);                // global_addr [95:64]
    g0[3] = (unsigned)((ga >> 32) & 0x01ffffffu)         // global_addr [120:96]
          | (2u << 30);                                  // type=2 ("image") [127:126]
    v8i_ g1;
    g1[0] = (int)((1u << 16)                             // data_size = 1 -> 2 bytes
          | (pad_en << 20)
          | (pad_interval_code << 22)
          | (pad_amount_code << 25));
    g1[1] = (int)((tensor_d0 & 0xffffu) << 16);          // tensor_dim0[15:0] @ bit48
    g1[2] = (int)((tensor_d0 >> 16) | ((tensor_d1 & 0xffffu) << 16));
    g1[3] = (int)((tensor_d1 >> 16) | (tile_d0 << 16));  // tile_dim0 @ [127:112]
    g1[4] = (int)(tile_d1);                              // tile_dim1 @ [143:128], tile_dim2=0
    g1[5] = (int)stride_d0;                              // tensor_dim0_stride[31:0]
    g1[6] = 0;
    g1[7] = 0;
    v4i_ gz4 = {0, 0, 0, 0};
    v8i_ gz8 = {0, 0, 0, 0, 0, 0, 0, 0};
    // amdgpu-toolchain (clang-23) 6-arg form
    __builtin_amdgcn_tensor_load_to_lds(g0, g1, gz4, gz4, gz8, 0);
}

// ---------------- Kernel 0: spectral-norm sigmas ----------------
__global__ void sigma_kernel(const float* __restrict__ wf, const float* __restrict__ uf,
                             const float* __restrict__ wg, const float* __restrict__ ug,
                             const float* __restrict__ wh, const float* __restrict__ uh,
                             const float* __restrict__ wo, const float* __restrict__ uo,
                             float* __restrict__ sig) {
    __shared__ float vsh[256];
    __shared__ float red[256];
    const int which = blockIdx.x;
    const float* w; const float* u; int in, out;
    switch (which) {
        case 0:  w = wf; u = uf; in = 256; out = 32;  break;
        case 1:  w = wg; u = ug; in = 256; out = 32;  break;
        case 2:  w = wh; u = uh; in = 256; out = 128; break;
        default: w = wo; u = uo; in = 128; out = 256; break;
    }
    const int t = threadIdx.x;
    float vr = 0.f;
    if (t < in) for (int j = 0; j < out; ++j) vr += u[j] * w[t*out + j];
    vsh[t] = (t < in) ? vr : 0.f;
    red[t] = (t < in) ? vr*vr : 0.f;
    __syncthreads();
    for (int s = 128; s > 0; s >>= 1) { if (t < s) red[t] += red[t+s]; __syncthreads(); }
    const float inv = rsqrtf(red[0] + 1e-12f);
    __syncthreads();
    float tr = 0.f;
    if (t < out) for (int i = 0; i < in; ++i) tr += vsh[i]*inv * w[i*out + t];
    red[t] = (t < out) ? tr*tr : 0.f;
    __syncthreads();
    for (int s = 128; s > 0; s >>= 1) { if (t < s) red[t] += red[t+s]; __syncthreads(); }
    if (t == 0) {
        float s2 = red[0];
        sig[which] = s2 * rsqrtf(s2 + 1e-12f);
    }
}

// ---------------- Kernel 1: normalized weights -> bf16, pre-swizzled to frag order ----
__global__ void wprep_kernel(const float* __restrict__ wf, const float* __restrict__ wg,
                             const float* __restrict__ wh, const float* __restrict__ wo,
                             const float* __restrict__ sig,
                             u16t* __restrict__ wcat_sw, u16t* __restrict__ wo_sw) {
    int idx = blockIdx.x * 256 + threadIdx.x;
    if (idx < 96*512) {
        int bi = idx >> 9, r = idx & 511, lane = r >> 4, e = r & 15;
        int ks = bi / 12, nt = bi % 12;
        int col = nt*16 + (lane & 15);
        int k   = 32*ks + (e & 7) + 8*(lane >> 4) + 16*(e >> 3);
        float v;
        if (col < 32)      v = wf[k*32  +  col      ] / sig[0];
        else if (col < 64) v = wg[k*32  + (col - 32)] / sig[1];
        else               v = wh[k*128 + (col - 64)] / sig[2];
        wcat_sw[idx] = f2bfbits(v);
    } else if (idx < 96*512 + 64*512) {
        int j = idx - 96*512;
        int bi = j >> 9, r = j & 511, lane = r >> 4, e = r & 15;
        int ks = bi >> 4, nt = bi & 15;
        int col = nt*16 + (lane & 15);
        int k   = 32*ks + (e & 7) + 8*(lane >> 4) + 16*(e >> 3);
        wo_sw[j] = f2bfbits(wo[k*256 + col] / sig[3]);
    }
}

// ---------------- Kernel 2: projections via bf16 WMMA ----------------
__global__ void __launch_bounds__(128) proj_kernel(const float* __restrict__ x,
                                                   const u16t* __restrict__ wcat_sw,
                                                   const float* __restrict__ bfp,
                                                   const float* __restrict__ bgp,
                                                   const float* __restrict__ bhp,
                                                   u16t* __restrict__ fg,
                                                   u16t* __restrict__ hT) {
    const int lane = threadIdx.x & 31;
    const int wave = threadIdx.x >> 5;
    const int mtile = blockIdx.x * 4 + wave;
    const int m = lane & 15, hi = lane >> 4;
    const int row = mtile*16 + m;

    v8f vz = {};
    v8f acc[12];
#pragma unroll
    for (int t = 0; t < 12; ++t) acc[t] = vz;

    for (int ks = 0; ks < 8; ++ks) {
        v16bf a;
        const float4* xr = (const float4*)(x + (size_t)row*CIN + 32*ks + 8*hi);
        float4 f0 = xr[0], f1 = xr[1], f2 = xr[4], f3 = xr[5];
        a[0]=f2bf(f0.x); a[1]=f2bf(f0.y); a[2]=f2bf(f0.z); a[3]=f2bf(f0.w);
        a[4]=f2bf(f1.x); a[5]=f2bf(f1.y); a[6]=f2bf(f1.z); a[7]=f2bf(f1.w);
        a[8]=f2bf(f2.x); a[9]=f2bf(f2.y); a[10]=f2bf(f2.z); a[11]=f2bf(f2.w);
        a[12]=f2bf(f3.x); a[13]=f2bf(f3.y); a[14]=f2bf(f3.z); a[15]=f2bf(f3.w);
#pragma unroll
        for (int nt = 0; nt < 12; ++nt) {
            v16bf bm = load_frag32(wcat_sw + ((size_t)((ks*12 + nt)*32 + lane))*16);
            acc[nt] = __builtin_amdgcn_wmma_f32_16x16x32_bf16(
                false, a, false, bm, (short)0, acc[nt], false, false);
        }
    }
#pragma unroll
    for (int nt = 0; nt < 4; ++nt) {
        int col = nt*16 + m;
        float bias = (col < 32) ? bfp[col] : bgp[col-32];
#pragma unroll
        for (int r = 0; r < 8; ++r) {
            int grow = mtile*16 + r + 8*hi;
            fg[(size_t)grow*64 + col] = f2bfbits(acc[nt][r] + bias);
        }
    }
    const int bidx = mtile >> 8;
    const int key0 = (mtile & 255)*16 + 8*hi;
#pragma unroll
    for (int nt = 4; nt < 12; ++nt) {
        int c = nt*16 + m - 64;
        float bias = bhp[c];
        u16t tmp[8];
#pragma unroll
        for (int r = 0; r < 8; ++r) tmp[r] = f2bfbits(acc[nt][r] + bias);
        uint4 v; __builtin_memcpy(&v, tmp, 16);
        *(uint4*)(hT + ((size_t)(bidx*128 + c))*HWN + key0) = v;
    }
}

// ---------------- Kernel 3: flash attention (TDM-staged chunks) ----------------
__global__ void __launch_bounds__(256) attn_kernel(const u16t* __restrict__ fg,
                                                   const u16t* __restrict__ hT,
                                                   u16t* __restrict__ obf) {
    __shared__ __align__(16) u16t fS[128*32];        // f chunk, row-major  (8 KB)
    __shared__ __align__(16) u16t hS[128*136];       // h chunk [col][key], pad 136 (34 KB)
    __shared__ __align__(16) u16t pS[8][2][512];     // per-wave double-buffered P (16 KB)

    const int tid = threadIdx.x;
    const int lane = tid & 31, wave = tid >> 5;
    const int m = lane & 15, hi = lane >> 4;
    const int b = blockIdx.y;
    const int q0 = blockIdx.x * 128 + wave * 16;
    const size_t brow = (size_t)b * HWN;

    v16bf ga;
    {
        const u16t* gp = fg + (brow + q0 + m)*64 + 32;
        uint4 a0 = *(const uint4*)(gp + 16*hi);
        uint4 a1 = *(const uint4*)(gp + 16*hi + 8);
        __builtin_memcpy(&ga, &a0, 16);
        __builtin_memcpy(((char*)&ga) + 16, &a1, 16);
    }

    float m_run[8], l_run[8];
    v8f vz = {};
    v8f oacc[8];
#pragma unroll
    for (int r = 0; r < 8; ++r) { m_run[r] = -1e30f; l_run[r] = 0.f; }
#pragma unroll
    for (int t = 0; t < 8; ++t) oacc[t] = vz;

    const unsigned fS_lds = (unsigned)(size_t)(void*)fS;   // LDS aperture: addr[31:0]
    const unsigned hS_lds = (unsigned)(size_t)(void*)hS;

    for (int ch = 0; ch < 32; ++ch) {
        __syncthreads();                            // prev chunk reads complete
        if (wave == 0) {                            // one TDM issue per chunk (uniform branch)
            const size_t kg = brow + (size_t)ch * 128;
            // f chunk: 128 key-rows x 32 ch, global stride 64, packed in LDS
            tdm_load_2d(fS_lds, fg + kg*64, /*tile*/32, 128,
                        /*tensor*/64, ROWS, /*stride*/64, 0, 0, 0);
            // h chunk: 128 col-rows x 128 keys, global stride 4096,
            // LDS pad 4 DWORDs every 64 DWORDs -> row stride 136 u16
            tdm_load_2d(hS_lds, hT + ((size_t)(b*128))*HWN + ch*128, /*tile*/128, 128,
                        /*tensor*/HWN, 1024, /*stride*/HWN, 1, 5, 3);
            __builtin_amdgcn_s_wait_tensorcnt(0);
        }
        __syncthreads();                            // staged data visible to all waves

        for (int jj = 0; jj < 4; ++jj) {            // 32 keys per step
            const int kk = jj * 32;
            v16bf bf0 = load_frag32(fS + (kk + m)*32 + 16*hi);
            v16bf bf1 = load_frag32(fS + (kk + 16 + m)*32 + 16*hi);
            v8f s0 = __builtin_amdgcn_wmma_f32_16x16x32_bf16(
                false, ga, false, bf0, (short)0, vz, false, false);
            v8f s1 = __builtin_amdgcn_wmma_f32_16x16x32_bf16(
                false, ga, false, bf1, (short)0, vz, false, false);

            float mx[8];
#pragma unroll
            for (int r = 0; r < 8; ++r) mx[r] = fmaxf(s0[r], s1[r]);
#pragma unroll
            for (int off = 1; off < 16; off <<= 1)
#pragma unroll
                for (int r = 0; r < 8; ++r)
                    mx[r] = fmaxf(mx[r], __shfl_xor(mx[r], off, 32));

            float p0v[8], p1v[8], rs[8];
#pragma unroll
            for (int r = 0; r < 8; ++r) {
                float mn = fmaxf(m_run[r], mx[r]);
                float alpha = __expf(m_run[r] - mn);
                m_run[r] = mn;
                l_run[r] *= alpha;
#pragma unroll
                for (int nt = 0; nt < 8; ++nt) oacc[nt][r] *= alpha;
                p0v[r] = __expf(s0[r] - mn);
                p1v[r] = __expf(s1[r] - mn);
                rs[r] = p0v[r] + p1v[r];
            }
#pragma unroll
            for (int off = 1; off < 16; off <<= 1)
#pragma unroll
                for (int r = 0; r < 8; ++r) rs[r] += __shfl_xor(rs[r], off, 32);
#pragma unroll
            for (int r = 0; r < 8; ++r) l_run[r] += rs[r];

            // double-buffered P staging: WAR on same buffer is 2 barriers away
            u16t* pw = &pS[wave][jj & 1][0];
#pragma unroll
            for (int r = 0; r < 8; ++r) {
                int prow = r + 8*hi;
                pw[prow*32 + m]      = f2bfbits(p0v[r]);
                pw[prow*32 + 16 + m] = f2bfbits(p1v[r]);
            }
            __syncthreads();
            v16bf pa;
            {
                const u16t* pp = pw + m*32;
                uint4 a0 = *(const uint4*)(pp + 16*hi);
                uint4 a1 = *(const uint4*)(pp + 16*hi + 8);
                __builtin_memcpy(&pa, &a0, 16);
                __builtin_memcpy(((char*)&pa) + 16, &a1, 16);
            }
#pragma unroll
            for (int nt = 0; nt < 8; ++nt) {
                v16bf bh = load_frag32(hS + (nt*16 + m)*136 + kk + 16*hi);
                oacc[nt] = __builtin_amdgcn_wmma_f32_16x16x32_bf16(
                    false, pa, false, bh, (short)0, oacc[nt], false, false);
            }
        }
    }
#pragma unroll
    for (int r = 0; r < 8; ++r) {
        float inv = 1.0f / l_run[r];
#pragma unroll
        for (int nt = 0; nt < 8; ++nt) oacc[nt][r] *= inv;
    }
#pragma unroll
    for (int nt = 0; nt < 8; ++nt) {
        int col = nt*16 + m;
#pragma unroll
        for (int r = 0; r < 8; ++r) {
            size_t grow = brow + q0 + r + 8*hi;
            obf[grow*CH_ + col] = f2bfbits(oacc[nt][r]);
        }
    }
}

// ---------------- Kernel 4: o @ wo + bias, out = gamma*o + x ----------------
__global__ void __launch_bounds__(128) outproj_kernel(const u16t* __restrict__ obf,
                                                      const u16t* __restrict__ wo_sw,
                                                      const float* __restrict__ bo,
                                                      const float* __restrict__ x,
                                                      const float* __restrict__ gamma,
                                                      float* __restrict__ out) {
    const int lane = threadIdx.x & 31, wave = threadIdx.x >> 5;
    const int m = lane & 15, hi = lane >> 4;
    const int mtile = blockIdx.x;
    const int row = mtile*16 + m;

    v8f vz = {};
    v8f acc[4];
#pragma unroll
    for (int t = 0; t < 4; ++t) acc[t] = vz;

    for (int ks = 0; ks < 4; ++ks) {
        v16bf a;
        {
            const u16t* ap = obf + (size_t)row*CH_ + 32*ks;
            uint4 a0 = *(const uint4*)(ap + 8*hi);
            uint4 a1 = *(const uint4*)(ap + 16 + 8*hi);
            __builtin_memcpy(&a, &a0, 16);
            __builtin_memcpy(((char*)&a) + 16, &a1, 16);
        }
#pragma unroll
        for (int t = 0; t < 4; ++t) {
            int nt = wave*4 + t;
            v16bf bm = load_frag32(wo_sw + ((size_t)((ks*16 + nt)*32 + lane))*16);
            acc[t] = __builtin_amdgcn_wmma_f32_16x16x32_bf16(
                false, a, false, bm, (short)0, acc[t], false, false);
        }
    }
    const float g = gamma[0];
#pragma unroll
    for (int t = 0; t < 4; ++t) {
        int col = (wave*4 + t)*16 + m;
        float bias = bo[col];
#pragma unroll
        for (int r = 0; r < 8; ++r) {
            size_t grow = (size_t)mtile*16 + r + 8*hi;
            out[grow*CIN + col] = g * (acc[t][r] + bias) + x[grow*CIN + col];
        }
    }
}

// ---------------- launcher ----------------
extern "C" void kernel_launch(void* const* d_in, const int* in_sizes, int n_in,
                              void* d_out, int out_size, void* d_ws, size_t ws_size,
                              hipStream_t stream) {
    const float* x     = (const float*)d_in[0];
    const float* wf    = (const float*)d_in[1];
    const float* bfp   = (const float*)d_in[2];
    const float* uf    = (const float*)d_in[3];
    const float* wg    = (const float*)d_in[4];
    const float* bgp   = (const float*)d_in[5];
    const float* ug    = (const float*)d_in[6];
    const float* wh    = (const float*)d_in[7];
    const float* bhp   = (const float*)d_in[8];
    const float* uh    = (const float*)d_in[9];
    const float* wo    = (const float*)d_in[10];
    const float* bo    = (const float*)d_in[11];
    const float* uo    = (const float*)d_in[12];
    const float* gamma = (const float*)d_in[13];
    float* out = (float*)d_out;

    char* ws = (char*)d_ws;
    float* sig     = (float*)(ws + OFF_SIG);
    u16t* wcat_sw  = (u16t*)(ws + OFF_WCAT);
    u16t* wo_sw    = (u16t*)(ws + OFF_WO);
    u16t* fg       = (u16t*)(ws + OFF_FG);
    u16t* hT       = (u16t*)(ws + OFF_HT);
    u16t* obf      = (u16t*)(ws + OFF_OBF);

    sigma_kernel<<<4, 256, 0, stream>>>(wf, uf, wg, ug, wh, uh, wo, uo, sig);
    wprep_kernel<<<320, 256, 0, stream>>>(wf, wg, wh, wo, sig, wcat_sw, wo_sw);
    proj_kernel<<<512, 128, 0, stream>>>(x, wcat_sw, bfp, bgp, bhp, fg, hT);
    attn_kernel<<<dim3(32, 8), 256, 0, stream>>>(fg, hT, obf);
    outproj_kernel<<<2048, 128, 0, stream>>>(obf, wo_sw, bo, x, gamma, out);
}
